// ABCNN1Attention_11184094839544
// MI455X (gfx1250) — compile-verified, matrix operand order checked by reading, not run on hardware
//
#include <hip/hip_runtime.h>
#include <math.h>

// Problem constants (from the reference): B=64, L=512, D=1024
#define B_ 64
#define L_ 512
#define D_ 1024

typedef __attribute__((ext_vector_type(16))) _Float16 v16h;
typedef __attribute__((ext_vector_type(8)))  _Float16 v8h;
typedef __attribute__((ext_vector_type(4)))  _Float16 v4h;
typedef __attribute__((ext_vector_type(8)))  float    v8f;

// ---------------------------------------------------------------------------
// WMMA helpers (CDNA5 / gfx1250, wave32).
//
// 16-bit A-operand (16x32 MxK) per ISA 7.12.2:
//   lanes 0-15  : row M=lane,    K = {0..7, 16..23}
//   lanes 16-31 : row M=lane-16, K = {8..15, 24..31}
// B-operand (32x16 KxN) is the mirrored layout over columns, so loading a
// row-major B^T (N x K) uses the exact same pattern. Both chunks per lane are
// contiguous 16-byte runs -> two v8h vector loads.
// ---------------------------------------------------------------------------
__device__ __forceinline__ v16h load_frag16(const _Float16* __restrict__ base, int ld) {
  const int lane = threadIdx.x & 31;
  const int r    = lane & 15;
  const int hs   = lane >> 4;
  const _Float16* p = base + (size_t)r * ld + hs * 8;
  v8h lo = *(const v8h*)(p);
  v8h hi = *(const v8h*)(p + 16);
  v16h f;
#pragma unroll
  for (int i = 0; i < 8; ++i) { f[i] = lo[i]; f[i + 8] = hi[i]; }
  return f;
}

__device__ __forceinline__ v8f wmma_f16(v16h a, v16h b, v8f c) {
  // (neg_a, A, neg_b, B, c_mod, C, reuse_a, reuse_b)
  return __builtin_amdgcn_wmma_f32_16x16x32_f16(false, a, false, b, (short)0, c,
                                                false, false);
}

// ---------------------------------------------------------------------------
// Kernel 1: per-row prep. One 256-thread block per (tensor, row):
//   - copy fp32 row into output channel 0 (the concat with x)
//   - convert row to fp16 into workspace
//   - reduce sum-of-squares -> store reciprocal norm
// ---------------------------------------------------------------------------
__global__ void prep_kernel(const float* __restrict__ x1, const float* __restrict__ x2,
                            _Float16* __restrict__ x1h, _Float16* __restrict__ x2h,
                            float* __restrict__ rn1, float* __restrict__ rn2,
                            float* __restrict__ out) {
  __shared__ float red[256];
  const int row = blockIdx.x;          // 0 .. B*L-1
  const int sel = blockIdx.y;          // 0: x1 path, 1: x2 path
  const int b   = row / L_;
  const int l   = row % L_;

  const float* src = (sel ? x2 : x1) + (size_t)row * D_;
  _Float16*    dh  = (sel ? x2h : x1h) + (size_t)row * D_;
  float*       rdn = (sel ? rn2 : rn1);
  // attn1 base = out; attn2 base = out + B*2*L*D. Channel 0 holds the copy.
  float* cpy = out + (size_t)sel * ((size_t)B_ * 2 * L_ * D_)
                   + ((size_t)(2 * b) * L_ + l) * (size_t)D_;

  const int t = threadIdx.x;           // 256 threads * 4 floats = 1024 = D
  float4 v = ((const float4*)src)[t];
  ((float4*)cpy)[t] = v;
  v4h h;
  h[0] = (_Float16)v.x; h[1] = (_Float16)v.y;
  h[2] = (_Float16)v.z; h[3] = (_Float16)v.w;
  ((v4h*)dh)[t] = h;

  red[t] = v.x * v.x + v.y * v.y + v.z * v.z + v.w * v.w;
  __syncthreads();
#pragma unroll
  for (int s = 128; s > 0; s >>= 1) {
    if (t < s) red[t] += red[t + s];
    __syncthreads();
  }
  if (t == 0) rdn[row] = 1.0f / sqrtf(red[0]);
}

// ---------------------------------------------------------------------------
// Kernel 2: W (L x D, fp32) -> W^T (D x L, fp16). Tiny (2 MiB total), simple.
// ---------------------------------------------------------------------------
__global__ void wt_kernel(const float* __restrict__ W1, const float* __restrict__ W2,
                          _Float16* __restrict__ W1t, _Float16* __restrict__ W2t) {
  const int idx = blockIdx.x * 256 + threadIdx.x;   // 0 .. L*D-1
  const float* W  = blockIdx.y ? W2 : W1;
  _Float16*   Wt  = blockIdx.y ? W2t : W1t;
  const int l = idx / D_;
  const int d = idx % D_;
  Wt[(size_t)d * L_ + l] = (_Float16)W[idx];
}

// ---------------------------------------------------------------------------
// Kernel 3: cosine score matrix. One wave computes a 32x32 tile of A for one
// batch (2x2 sub-tiles, 4 fp32 accumulators). K-loop over D=1024, unrolled by
// 2 K-steps with ping-pong fragment sets: wmma(set0) -> reload set0 ->
// wmma(set1) -> reload set1. Operands are overwritten in place, so the
// double-buffer hand-off costs zero v_mov instructions, and each set's loads
// overlap the other set's WMMAs.
// Applies 1/(|x1_l||x2_m|) in registers; stores A and A^T as fp16.
// ---------------------------------------------------------------------------
__global__ void __launch_bounds__(32, 1)
score_kernel(const _Float16* __restrict__ x1h,
             const _Float16* __restrict__ x2h,
             const float* __restrict__ rn1,
             const float* __restrict__ rn2,
             _Float16* __restrict__ Ah,
             _Float16* __restrict__ Ath) {
  const int b    = blockIdx.z;
  const int row0 = blockIdx.x * 32;   // l-tile (2 sub-tiles of 16)
  const int col0 = blockIdx.y * 32;   // m-tile (2 sub-tiles of 16)
  const _Float16* x1p = x1h + (size_t)b * L_ * D_ + (size_t)row0 * D_;
  const _Float16* x2p = x2h + (size_t)b * L_ * D_ + (size_t)col0 * D_;

  v8f acc[2][2] = {};
  v16h a0[2], b0[2], a1[2], b1[2];
#pragma unroll
  for (int m = 0; m < 2; ++m) a0[m] = load_frag16(x1p + (size_t)(m * 16) * D_, D_);
#pragma unroll
  for (int j = 0; j < 2; ++j) b0[j] = load_frag16(x2p + (size_t)(j * 16) * D_, D_);
#pragma unroll
  for (int m = 0; m < 2; ++m) a1[m] = load_frag16(x1p + (size_t)(m * 16) * D_ + 32, D_);
#pragma unroll
  for (int j = 0; j < 2; ++j) b1[j] = load_frag16(x2p + (size_t)(j * 16) * D_ + 32, D_);

  for (int k0 = 64; k0 < D_; k0 += 64) {
#pragma unroll
    for (int m = 0; m < 2; ++m)
#pragma unroll
      for (int j = 0; j < 2; ++j)
        acc[m][j] = wmma_f16(a0[m], b0[j], acc[m][j]);
#pragma unroll
    for (int m = 0; m < 2; ++m)
      a0[m] = load_frag16(x1p + (size_t)(m * 16) * D_ + k0, D_);
#pragma unroll
    for (int j = 0; j < 2; ++j)
      b0[j] = load_frag16(x2p + (size_t)(j * 16) * D_ + k0, D_);
#pragma unroll
    for (int m = 0; m < 2; ++m)
#pragma unroll
      for (int j = 0; j < 2; ++j)
        acc[m][j] = wmma_f16(a1[m], b1[j], acc[m][j]);
#pragma unroll
    for (int m = 0; m < 2; ++m)
      a1[m] = load_frag16(x1p + (size_t)(m * 16) * D_ + k0 + 32, D_);
#pragma unroll
    for (int j = 0; j < 2; ++j)
      b1[j] = load_frag16(x2p + (size_t)(j * 16) * D_ + k0 + 32, D_);
  }
#pragma unroll
  for (int m = 0; m < 2; ++m)
#pragma unroll
    for (int j = 0; j < 2; ++j)
      acc[m][j] = wmma_f16(a0[m], b0[j], acc[m][j]);
#pragma unroll
  for (int m = 0; m < 2; ++m)
#pragma unroll
    for (int j = 0; j < 2; ++j)
      acc[m][j] = wmma_f16(a1[m], b1[j], acc[m][j]);

  const int lane = threadIdx.x & 31;
  const int r  = lane & 15;
  const int hs = lane >> 4;
  _Float16* Ab  = Ah  + (size_t)b * L_ * L_;
  _Float16* Atb = Ath + (size_t)b * L_ * L_;
#pragma unroll
  for (int m = 0; m < 2; ++m) {
    float r1v[8];
#pragma unroll
    for (int i = 0; i < 8; ++i)
      r1v[i] = rn1[b * L_ + row0 + m * 16 + hs * 8 + i];
#pragma unroll
    for (int j = 0; j < 2; ++j) {
      const int gn = col0 + j * 16 + r;
      const float r2 = rn2[b * L_ + gn];
#pragma unroll
      for (int i = 0; i < 8; ++i) {
        const int gm = row0 + m * 16 + hs * 8 + i;
        const _Float16 hv = (_Float16)(acc[m][j][i] * r1v[i] * r2);
        Ab [(size_t)gm * L_ + gn] = hv;
        Atb[(size_t)gn * L_ + gm] = hv;
      }
    }
  }
}

// ---------------------------------------------------------------------------
// Kernels 4/5: OUT(channel 1) = Amat (L x L, fp16) @ W (via W^T fp16, D x L).
// One wave per 32x32 output tile; K-loop over L=512 with the same ping-pong
// unroll-by-2 structure. fp32 stores direct to d_out.
// ---------------------------------------------------------------------------
__global__ void __launch_bounds__(32, 1)
gemm_out_kernel(const _Float16* __restrict__ Amat,
                const _Float16* __restrict__ Wt,
                float* __restrict__ outbase) {
  const int b    = blockIdx.z;
  const int row0 = blockIdx.y * 32;   // output row tile (l or m)
  const int col0 = blockIdx.x * 32;   // output col tile (d)
  const _Float16* Ap = Amat + (size_t)b * L_ * L_ + (size_t)row0 * L_;
  const _Float16* Bp = Wt + (size_t)col0 * L_;

  v8f acc[2][2] = {};
  v16h a0[2], b0[2], a1[2], b1[2];
#pragma unroll
  for (int m = 0; m < 2; ++m) a0[m] = load_frag16(Ap + (size_t)(m * 16) * L_, L_);
#pragma unroll
  for (int j = 0; j < 2; ++j) b0[j] = load_frag16(Bp + (size_t)(j * 16) * L_, L_);
#pragma unroll
  for (int m = 0; m < 2; ++m) a1[m] = load_frag16(Ap + (size_t)(m * 16) * L_ + 32, L_);
#pragma unroll
  for (int j = 0; j < 2; ++j) b1[j] = load_frag16(Bp + (size_t)(j * 16) * L_ + 32, L_);

  for (int k0 = 64; k0 < L_; k0 += 64) {
#pragma unroll
    for (int m = 0; m < 2; ++m)
#pragma unroll
      for (int j = 0; j < 2; ++j)
        acc[m][j] = wmma_f16(a0[m], b0[j], acc[m][j]);
#pragma unroll
    for (int m = 0; m < 2; ++m)
      a0[m] = load_frag16(Ap + (size_t)(m * 16) * L_ + k0, L_);
#pragma unroll
    for (int j = 0; j < 2; ++j)
      b0[j] = load_frag16(Bp + (size_t)(j * 16) * L_ + k0, L_);
#pragma unroll
    for (int m = 0; m < 2; ++m)
#pragma unroll
      for (int j = 0; j < 2; ++j)
        acc[m][j] = wmma_f16(a1[m], b1[j], acc[m][j]);
#pragma unroll
    for (int m = 0; m < 2; ++m)
      a1[m] = load_frag16(Ap + (size_t)(m * 16) * L_ + k0 + 32, L_);
#pragma unroll
    for (int j = 0; j < 2; ++j)
      b1[j] = load_frag16(Bp + (size_t)(j * 16) * L_ + k0 + 32, L_);
  }
#pragma unroll
  for (int m = 0; m < 2; ++m)
#pragma unroll
    for (int j = 0; j < 2; ++j)
      acc[m][j] = wmma_f16(a0[m], b0[j], acc[m][j]);
#pragma unroll
  for (int m = 0; m < 2; ++m)
#pragma unroll
    for (int j = 0; j < 2; ++j)
      acc[m][j] = wmma_f16(a1[m], b1[j], acc[m][j]);

  const int lane = threadIdx.x & 31;
  const int r  = lane & 15;
  const int hs = lane >> 4;
  float* op = outbase + ((size_t)(2 * b + 1) * L_) * (size_t)D_;  // channel 1
#pragma unroll
  for (int m = 0; m < 2; ++m)
#pragma unroll
    for (int j = 0; j < 2; ++j)
#pragma unroll
      for (int i = 0; i < 8; ++i)
        op[(size_t)(row0 + m * 16 + hs * 8 + i) * D_ + (col0 + j * 16 + r)] =
            acc[m][j][i];
}

// ---------------------------------------------------------------------------
extern "C" void kernel_launch(void* const* d_in, const int* in_sizes, int n_in,
                              void* d_out, int out_size, void* d_ws, size_t ws_size,
                              hipStream_t stream) {
  const float* x1 = (const float*)d_in[0];
  const float* x2 = (const float*)d_in[1];
  const float* W1 = (const float*)d_in[2];
  const float* W2 = (const float*)d_in[3];
  float* out = (float*)d_out;

  // Workspace carve (~195 MiB), 256B-aligned regions.
  char* ws = (char*)d_ws;
  size_t off = 0;
  auto carve = [&](size_t bytes) -> char* {
    char* p = ws + off;
    off += (bytes + 255) & ~(size_t)255;
    return p;
  };
  _Float16* x1h = (_Float16*)carve((size_t)B_ * L_ * D_ * 2);
  _Float16* x2h = (_Float16*)carve((size_t)B_ * L_ * D_ * 2);
  _Float16* Ah  = (_Float16*)carve((size_t)B_ * L_ * L_ * 2);
  _Float16* Ath = (_Float16*)carve((size_t)B_ * L_ * L_ * 2);
  _Float16* W1t = (_Float16*)carve((size_t)L_ * D_ * 2);
  _Float16* W2t = (_Float16*)carve((size_t)L_ * D_ * 2);
  float*    rn1 = (float*)carve((size_t)B_ * L_ * 4);
  float*    rn2 = (float*)carve((size_t)B_ * L_ * 4);

  // 1) copies into channel 0 + fp16 conversion + reciprocal norms
  prep_kernel<<<dim3(B_ * L_, 2), 256, 0, stream>>>(x1, x2, x1h, x2h, rn1, rn2, out);
  // 2) weight transpose + fp16
  wt_kernel<<<dim3((L_ * D_) / 256, 2), 256, 0, stream>>>(W1, W2, W1t, W2t);
  // 3) cosine attention matrix A and A^T (fp16)
  score_kernel<<<dim3(L_ / 32, L_ / 32, B_), 32, 0, stream>>>(x1h, x2h, rn1, rn2, Ah, Ath);
  // 4) a1 = A @ W1  -> attn1 channel 1
  gemm_out_kernel<<<dim3(D_ / 32, L_ / 32, B_), 32, 0, stream>>>(Ah, W1t, out);
  // 5) a2 = A^T @ W2 -> attn2 channel 1
  gemm_out_kernel<<<dim3(D_ / 32, L_ / 32, B_), 32, 0, stream>>>(
      Ath, W2t, out + (size_t)B_ * 2 * L_ * D_);
}
